// MixtralDecoderLayer_14955076125240
// MI455X (gfx1250) — matrix operand order repaired and checked by reading, not compiled, not run on hardware
//
#include <hip/hip_runtime.h>
#include <hip/hip_bf16.h>
#include <cstdint>
#include <cstddef>

#define B_      2
#define S_      1024
#define H_      1024
#define NH_     16
#define NKV_    8
#define HD_     64
#define F_      3584
#define E_      8
#define WINDOW_ 512
#define BS_     (B_ * S_)
#define GROWS_  (2 * BS_)   // TOPK=2 -> exactly 2 gathered rows per token

typedef __attribute__((ext_vector_type(16))) __bf16 bf16x16;
typedef __attribute__((ext_vector_type(8)))  float  f32x8;

union FragU    { bf16x16 v; unsigned int u[8]; };
union Bf16Bits { unsigned short u; __bf16 b; };

__device__ __forceinline__ __bf16 f2bf(float f) {
  union { float f; unsigned int u; } x; x.f = f;
  unsigned int r = x.u + 0x7FFFu + ((x.u >> 16) & 1u);   // round-to-nearest-even
  Bf16Bits o; o.u = (unsigned short)(r >> 16);
  return o.b;
}

__device__ __forceinline__ f32x8 zero8() {
  f32x8 z;
  #pragma unroll
  for (int i = 0; i < 8; ++i) z[i] = 0.f;
  return z;
}

// gfx1250 async global->LDS copy (ASYNCcnt), 16B per lane.
__device__ __forceinline__ void async_b128(const void* smem_dst, const void* gsrc) {
  unsigned lo = (unsigned)(size_t)smem_dst;             // LDS offset = low 32 bits
  unsigned long long ga = (unsigned long long)gsrc;
  asm volatile("global_load_async_to_lds_b128 %0, %1, off"
               :: "v"(lo), "v"(ga) : "memory");
}
__device__ __forceinline__ void wait_async0() {
  asm volatile("s_wait_asynccnt 0x0" ::: "memory");
}

// ---- WMMA fragment loaders (ISA 7.12.2, wave32, bf16 16x16x32) -------------
// A fragment 16x32 (MxK). smem layout: [row][k], rowStride elems, k pairs contiguous.
__device__ __forceinline__ bf16x16 frag_a(const __bf16* smem, int rowStride,
                                          int mBase, int kBase, int lane) {
  FragU f;
  const int m = mBase + (lane & 15);
  const int h = (lane >> 4) * 8;
  const unsigned int* p = (const unsigned int*)(smem + (size_t)m * rowStride);
  #pragma unroll
  for (int r = 0; r < 4; ++r) f.u[r]     = p[(kBase + h + 2 * r) >> 1];
  #pragma unroll
  for (int r = 0; r < 4; ++r) f.u[4 + r] = p[(kBase + 16 + h + 2 * r) >> 1];
  return f.v;
}

// B fragment 32x16 (KxN). smem layout: [n][k] (transposed), k pairs contiguous.
__device__ __forceinline__ bf16x16 frag_b(const __bf16* smem, int rowStride,
                                          int nBase, int kBase, int lane) {
  FragU f;
  const int n  = nBase + (lane & 15);
  const int kb = kBase + (lane >> 4) * 8;
  const unsigned int* p = (const unsigned int*)(smem + (size_t)n * rowStride);
  #pragma unroll
  for (int r = 0; r < 8; ++r) f.u[r] = p[(kb + 2 * r + (r >= 4 ? 8 : 0)) >> 1];
  return f.v;
}

// ---- Generic bf16 WMMA GEMM: C[M,N] = A[M,K] * B[K,N] (+epilogue) ----------
// Double-buffered LDS; A via async global->LDS, B software-pipelined + transposed.
__global__ __launch_bounds__(256) void gemm_bf16_wmma(
    const __bf16* __restrict__ A, int lda,
    const __bf16* __restrict__ Bm, int ldb,
    float* __restrict__ C, int ldc,
    int M, int N, int K,
    const int* __restrict__ seg, int expert,
    const float* __restrict__ residual,
    const float* __restrict__ row_scale,
    const int* __restrict__ row_map,
    int accumulate) {
  __shared__ __attribute__((aligned(16))) __bf16 As[2][128 * 64];  // [m][k]
  __shared__ __attribute__((aligned(16))) __bf16 Bs[2][128 * 64];  // [n][k]

  int rowbase = 0, Me = M;
  if (seg) { rowbase = seg[expert]; Me = seg[expert + 1] - rowbase; }
  const int m0 = blockIdx.y * 128;
  if (m0 >= Me) return;
  const int n0   = blockIdx.x * 128;
  const int tid  = threadIdx.x;
  const int lane = tid & 31;
  const int wave = tid >> 5;
  const int wm   = (wave >> 1) * 32;  // 4 waves along M
  const int wn   = (wave & 1) * 64;   // 2 waves along N

  // A slab -> LDS via async copies (stale data in masked rows is discarded
  // by the epilogue row guard, so no zero-fill needed).
  auto issue_A_async = [&](int k0, __bf16* dst) {
    #pragma unroll
    for (int i = 0; i < 4; ++i) {
      int c = tid + 256 * i;
      int r = c >> 3, kc = (c & 7) * 8;
      int gm = m0 + r;
      if (gm < Me)
        async_b128(dst + r * 64 + kc,
                   A + (size_t)(rowbase + gm) * lda + k0 + kc);
    }
  };
  auto load_B_regs = [&](int k0, uint4* regs) {
    #pragma unroll
    for (int i = 0; i < 4; ++i) {
      int c = tid + 256 * i;
      int kr = c >> 4, nc = (c & 15) * 8;
      regs[i] = *(const uint4*)(Bm + (size_t)(k0 + kr) * ldb + n0 + nc);
    }
  };
  auto store_B_lds = [&](const uint4* regs, __bf16* dst) {
    #pragma unroll
    for (int i = 0; i < 4; ++i) {
      int c = tid + 256 * i;
      int kr = c >> 4, nc = (c & 15) * 8;
      const __bf16* dv = (const __bf16*)&regs[i];
      #pragma unroll
      for (int j = 0; j < 8; ++j) dst[(size_t)(nc + j) * 64 + kr] = dv[j];
    }
  };

  f32x8 acc[2][4];
  #pragma unroll
  for (int i = 0; i < 2; ++i)
    #pragma unroll
    for (int j = 0; j < 4; ++j) acc[i][j] = zero8();

  const int nk = K / 64;
  {  // prologue: slab 0 into buffer 0
    uint4 br[4];
    issue_A_async(0, As[0]);
    load_B_regs(0, br);
    store_B_lds(br, Bs[0]);
    wait_async0();
  }
  __syncthreads();

  for (int ks = 0; ks < nk; ++ks) {
    const int cur = ks & 1, nxt = cur ^ 1;
    const bool more = (ks + 1 < nk);
    uint4 br[4];
    if (more) {  // overlap next-slab fetch with this slab's WMMAs
      issue_A_async((ks + 1) * 64, As[nxt]);
      load_B_regs((ks + 1) * 64, br);
      __builtin_prefetch(Bm + (size_t)((ks + 1) * 64 + (tid & 63)) * ldb + n0, 0, 1);
    }
    #pragma unroll
    for (int kk = 0; kk < 64; kk += 32) {
      bf16x16 a0 = frag_a(As[cur], 64, wm,      kk, lane);
      bf16x16 a1 = frag_a(As[cur], 64, wm + 16, kk, lane);
      #pragma unroll
      for (int nt = 0; nt < 4; ++nt) {
        bf16x16 b = frag_b(Bs[cur], 64, wn + nt * 16, kk, lane);
        acc[0][nt] = __builtin_amdgcn_wmma_f32_16x16x32_bf16(
            false, a0, false, b, (short)0, acc[0][nt], false, false);
        acc[1][nt] = __builtin_amdgcn_wmma_f32_16x16x32_bf16(
            false, a1, false, b, (short)0, acc[1][nt], false, false);
      }
    }
    if (more) store_B_lds(br, Bs[nxt]);
    wait_async0();
    __syncthreads();
  }

  const int half = lane >> 4, col = lane & 15;
  #pragma unroll
  for (int mt = 0; mt < 2; ++mt)
    #pragma unroll
    for (int nt = 0; nt < 4; ++nt)
      #pragma unroll
      for (int r = 0; r < 8; ++r) {
        int m = m0 + wm + mt * 16 + half * 8 + r;
        if (m >= Me) continue;
        int n = n0 + wn + nt * 16 + col;
        float v = acc[mt][nt][r];
        int grow = rowbase + m;
        if (row_scale) v *= row_scale[grow];
        size_t crow = row_map ? (size_t)row_map[grow] : (size_t)grow;
        float* cp = C + crow * ldc + n;
        if (residual) v += residual[crow * (size_t)ldc + n];
        if (accumulate) *cp += v; else *cp = v;
      }
}

// ---- RMSNorm -> bf16 (+ optional f32 copy) ---------------------------------
__global__ __launch_bounds__(256) void rmsnorm_kernel(
    const float* __restrict__ x, const float* __restrict__ w,
    __bf16* __restrict__ xb, float* __restrict__ xf) {
  const int t = blockIdx.x;
  const float* row = x + (size_t)t * H_;
  __shared__ float red[256];
  float s = 0.f;
  for (int i = threadIdx.x; i < H_; i += 256) { float v = row[i]; s += v * v; }
  red[threadIdx.x] = s;
  __syncthreads();
  for (int off = 128; off; off >>= 1) {
    if (threadIdx.x < off) red[threadIdx.x] += red[threadIdx.x + off];
    __syncthreads();
  }
  const float inv = rsqrtf(red[0] / (float)H_ + 1e-6f);
  for (int i = threadIdx.x; i < H_; i += 256) {
    float v = row[i] * inv * w[i];
    xb[(size_t)t * H_ + i] = f2bf(v);
    if (xf) xf[(size_t)t * H_ + i] = v;
  }
}

// ---- RoPE on q/k (f32 in), cast q/k/v -> bf16 ------------------------------
__global__ __launch_bounds__(256) void rope_cast_kernel(
    const float* __restrict__ qf, const float* __restrict__ kf,
    const float* __restrict__ vf, const int* __restrict__ pos_ids,
    __bf16* __restrict__ qb, __bf16* __restrict__ kb, __bf16* __restrict__ vb) {
  const int t = blockIdx.x;
  const float pos = (float)pos_ids[t];
  for (int i = threadIdx.x; i < NH_ * HD_; i += 256) {
    int d = i & (HD_ - 1), dd = d & (HD_ / 2 - 1);
    float ang = pos * __powf(10000.f, -(float)(2 * dd) / (float)HD_);
    float c = __cosf(ang), sn = __sinf(ang);
    float v0 = qf[(size_t)t * NH_ * HD_ + i];
    float ot = (d < HD_ / 2) ? -qf[(size_t)t * NH_ * HD_ + i + HD_ / 2]
                             :  qf[(size_t)t * NH_ * HD_ + i - HD_ / 2];
    qb[(size_t)t * NH_ * HD_ + i] = f2bf(v0 * c + ot * sn);
  }
  for (int i = threadIdx.x; i < NKV_ * HD_; i += 256) {
    int d = i & (HD_ - 1), dd = d & (HD_ / 2 - 1);
    float ang = pos * __powf(10000.f, -(float)(2 * dd) / (float)HD_);
    float c = __cosf(ang), sn = __sinf(ang);
    float v0 = kf[(size_t)t * NKV_ * HD_ + i];
    float ot = (d < HD_ / 2) ? -kf[(size_t)t * NKV_ * HD_ + i + HD_ / 2]
                             :  kf[(size_t)t * NKV_ * HD_ + i - HD_ / 2];
    kb[(size_t)t * NKV_ * HD_ + i] = f2bf(v0 * c + ot * sn);
    vb[(size_t)t * NKV_ * HD_ + i] = f2bf(vf[(size_t)t * NKV_ * HD_ + i]);
  }
}

// ---- Flash-style sliding-window GQA attention (WMMA bf16) ------------------
// grid: (S/64, NH, B); block: 128 (4 waves); each wave owns 16 query rows.
__global__ __launch_bounds__(128) void attn_kernel(
    const __bf16* __restrict__ qb, const __bf16* __restrict__ kb,
    const __bf16* __restrict__ vb, __bf16* __restrict__ ob) {
  const int qt = blockIdx.x, h = blockIdx.y, b = blockIdx.z;
  const int kvh = h >> 1;  // NH/NKV = 2
  __shared__ __attribute__((aligned(16))) __bf16 Qs[64 * 64];      // [q][d]
  __shared__ __attribute__((aligned(16))) __bf16 Ks[64 * 64];      // [k][d]
  __shared__ __attribute__((aligned(16))) __bf16 Vt[64 * 64];      // [d][k]
  __shared__ __attribute__((aligned(16))) __bf16 Ps[4 * 16 * 64];  // per-wave P
  const int tid = threadIdx.x, lane = tid & 31, wave = tid >> 5;
  const int half = lane >> 4, col = lane & 15;
  const int q0 = qt * 64;

  #pragma unroll
  for (int i = 0; i < 4; ++i) {  // Q tile: async global->LDS (layout-preserving)
    int c = tid + 128 * i, r = c >> 3, dc = (c & 7) * 8;
    async_b128(Qs + r * 64 + dc,
               qb + ((size_t)(b * S_ + q0 + r) * NH_ + h) * HD_ + dc);
  }

  f32x8 out[4];
  #pragma unroll
  for (int i = 0; i < 4; ++i) out[i] = zero8();
  float mrow[8], lrow[8];
  #pragma unroll
  for (int r = 0; r < 8; ++r) { mrow[r] = -1e30f; lrow[r] = 0.f; }

  int kstart = q0 - (WINDOW_ - 1);
  if (kstart < 0) kstart = 0;
  kstart &= ~63;

  for (int kt = kstart; kt < q0 + 64; kt += 64) {
    __syncthreads();
    #pragma unroll
    for (int i = 0; i < 4; ++i) {  // K tile [k][d]: async global->LDS
      int c = tid + 128 * i, r = c >> 3, dc = (c & 7) * 8;
      async_b128(Ks + r * 64 + dc,
                 kb + ((size_t)(b * S_ + kt + r) * NKV_ + kvh) * HD_ + dc);
    }
    #pragma unroll
    for (int i = 0; i < 4; ++i) {  // V tile transposed [d][k] via registers
      int c = tid + 128 * i, r = c >> 3, dc = (c & 7) * 8;
      uint4 d = *(const uint4*)(vb + ((size_t)(b * S_ + kt + r) * NKV_ + kvh) * HD_ + dc);
      const __bf16* dv = (const __bf16*)&d;
      #pragma unroll
      for (int j = 0; j < 8; ++j) Vt[(size_t)(dc + j) * 64 + r] = dv[j];
    }
    wait_async0();
    __syncthreads();

    f32x8 sc[4];
    #pragma unroll
    for (int i = 0; i < 4; ++i) sc[i] = zero8();
    #pragma unroll
    for (int kk = 0; kk < 64; kk += 32) {
      bf16x16 a = frag_a(Qs, 64, wave * 16, kk, lane);
      #pragma unroll
      for (int nt = 0; nt < 4; ++nt) {
        bf16x16 bfr = frag_b(Ks, 64, nt * 16, kk, lane);
        sc[nt] = __builtin_amdgcn_wmma_f32_16x16x32_bf16(
            false, a, false, bfr, (short)0, sc[nt], false, false);
      }
    }
    // mask + scale + row max
    float rmax[8];
    #pragma unroll
    for (int r = 0; r < 8; ++r) rmax[r] = -1e30f;
    #pragma unroll
    for (int nt = 0; nt < 4; ++nt)
      #pragma unroll
      for (int r = 0; r < 8; ++r) {
        int q = q0 + wave * 16 + half * 8 + r;
        int k = kt + nt * 16 + col;
        float s = sc[nt][r] * 0.125f;  // 1/sqrt(64)
        bool ok = (k <= q) && (q - k < WINDOW_);
        s = ok ? s : -3.0e38f;
        sc[nt][r] = s;
        rmax[r] = fmaxf(rmax[r], s);
      }
    #pragma unroll
    for (int m = 1; m < 16; m <<= 1)
      #pragma unroll
      for (int r = 0; r < 8; ++r)
        rmax[r] = fmaxf(rmax[r], __shfl_xor(rmax[r], m, 32));
    float alpha[8], rsum[8];
    #pragma unroll
    for (int r = 0; r < 8; ++r) {
      float mn = fmaxf(mrow[r], rmax[r]);
      alpha[r] = __expf(mrow[r] - mn);
      mrow[r] = mn;
      rsum[r] = 0.f;
    }
    #pragma unroll
    for (int nt = 0; nt < 4; ++nt)
      #pragma unroll
      for (int r = 0; r < 8; ++r) {
        float p = __expf(sc[nt][r] - mrow[r]);
        sc[nt][r] = p;
        rsum[r] += p;
      }
    #pragma unroll
    for (int m = 1; m < 16; m <<= 1)
      #pragma unroll
      for (int r = 0; r < 8; ++r) rsum[r] += __shfl_xor(rsum[r], m, 32);
    #pragma unroll
    for (int r = 0; r < 8; ++r) lrow[r] = lrow[r] * alpha[r] + rsum[r];
    #pragma unroll
    for (int dt = 0; dt < 4; ++dt)
      #pragma unroll
      for (int r = 0; r < 8; ++r) out[dt][r] *= alpha[r];

    // restripe P (C-layout) -> A-layout via per-wave LDS
    __bf16* pw = Ps + wave * 16 * 64;
    #pragma unroll
    for (int nt = 0; nt < 4; ++nt)
      #pragma unroll
      for (int r = 0; r < 8; ++r)
        pw[(half * 8 + r) * 64 + nt * 16 + col] = f2bf(sc[nt][r]);

    #pragma unroll
    for (int kk = 0; kk < 64; kk += 32) {
      bf16x16 a = frag_a(pw, 64, 0, kk, lane);
      #pragma unroll
      for (int dt = 0; dt < 4; ++dt) {
        bf16x16 bfr = frag_b(Vt, 64, dt * 16, kk, lane);
        out[dt] = __builtin_amdgcn_wmma_f32_16x16x32_bf16(
            false, a, false, bfr, (short)0, out[dt], false, false);
      }
    }
  }
  #pragma unroll
  for (int dt = 0; dt < 4; ++dt)
    #pragma unroll
    for (int r = 0; r < 8; ++r) {
      int q = q0 + wave * 16 + half * 8 + r;
      ob[((size_t)(b * S_ + q) * NH_ + h) * HD_ + dt * 16 + col] =
          f2bf(out[dt][r] / lrow[r]);
    }
}

// ---- Router GEMV (8 waves -> 8 experts) ------------------------------------
__global__ __launch_bounds__(256) void router_kernel(
    const float* __restrict__ y, const float* __restrict__ gw,
    float* __restrict__ logits) {
  const int t = blockIdx.x, lane = threadIdx.x & 31, e = threadIdx.x >> 5;
  float s = 0.f;
  for (int i = lane; i < H_; i += 32) s += y[(size_t)t * H_ + i] * gw[(size_t)i * E_ + e];
  #pragma unroll
  for (int m = 16; m >= 1; m >>= 1) s += __shfl_xor(s, m, 32);
  if (lane == 0) logits[(size_t)t * E_ + e] = s;
}

__global__ void zero_ints(int* p, int n) {
  int i = blockIdx.x * blockDim.x + threadIdx.x;
  if (i < n) p[i] = 0;
}

// ---- softmax + top-2 + per-expert counting ---------------------------------
__global__ __launch_bounds__(256) void topk_kernel(
    const float* __restrict__ logits, float* __restrict__ topw,
    int* __restrict__ topi, int* __restrict__ counts) {
  int t = blockIdx.x * blockDim.x + threadIdx.x;
  if (t >= BS_) return;
  float p[E_], mx = -3.0e38f;
  #pragma unroll
  for (int e = 0; e < E_; ++e) { p[e] = logits[(size_t)t * E_ + e]; mx = fmaxf(mx, p[e]); }
  float se = 0.f;
  #pragma unroll
  for (int e = 0; e < E_; ++e) { p[e] = __expf(p[e] - mx); se += p[e]; }
  #pragma unroll
  for (int e = 0; e < E_; ++e) p[e] /= se;
  int i0 = 0;
  #pragma unroll
  for (int e = 1; e < E_; ++e) if (p[e] > p[i0]) i0 = e;
  int i1 = -1;
  #pragma unroll
  for (int e = 0; e < E_; ++e) if (e != i0 && (i1 < 0 || p[e] > p[i1])) i1 = e;
  float ws = p[i0] + p[i1];
  topw[t * 2] = p[i0] / ws; topw[t * 2 + 1] = p[i1] / ws;
  topi[t * 2] = i0;         topi[t * 2 + 1] = i1;
  atomicAdd(&counts[i0], 1);
  atomicAdd(&counts[i1], 1);
}

__global__ void scan_kernel(const int* __restrict__ counts, int* __restrict__ seg) {
  if (blockIdx.x == 0 && threadIdx.x == 0) {
    int a = 0;
    for (int e = 0; e < E_; ++e) { seg[e] = a; a += counts[e]; }
    seg[E_] = a;  // == GROWS_
  }
}

__global__ __launch_bounds__(256) void scatter_kernel(
    const int* __restrict__ topi, const float* __restrict__ topw,
    const int* __restrict__ seg, int* __restrict__ cursor,
    int* __restrict__ row_map, float* __restrict__ row_scale,
    int* __restrict__ gsrc) {
  int t = blockIdx.x * blockDim.x + threadIdx.x;
  if (t >= BS_) return;
  #pragma unroll
  for (int s = 0; s < 2; ++s) {
    int e = topi[t * 2 + s];
    int pos = atomicAdd(&cursor[e], 1);
    int g = seg[e] + pos;
    row_map[g] = t * 2 + s;
    row_scale[g] = topw[t * 2 + s];
    gsrc[g] = t;
  }
}

__global__ __launch_bounds__(256) void gather_rows(
    const __bf16* __restrict__ yb, const int* __restrict__ gsrc,
    __bf16* __restrict__ ygb) {
  int g = blockIdx.x, t = gsrc[g];
  const unsigned int* src = (const unsigned int*)(yb + (size_t)t * H_);
  unsigned int* dst = (unsigned int*)(ygb + (size_t)g * H_);
  for (int i = threadIdx.x; i < H_ / 2; i += 256) dst[i] = src[i];
}

__global__ __launch_bounds__(256) void silu_mul_kernel(
    const float* __restrict__ g, const float* __restrict__ u,
    __bf16* __restrict__ tb, size_t n) {
  size_t i = (size_t)blockIdx.x * blockDim.x + threadIdx.x;
  size_t stride = (size_t)gridDim.x * blockDim.x;
  for (; i < n; i += stride) {
    float gv = g[i];
    float s = gv / (1.f + __expf(-gv));
    tb[i] = f2bf(s * u[i]);
  }
}

__global__ __launch_bounds__(256) void cast_f32_bf16(
    const float* __restrict__ src, __bf16* __restrict__ dst, size_t n) {
  size_t i = (size_t)blockIdx.x * blockDim.x + threadIdx.x;
  size_t stride = (size_t)gridDim.x * blockDim.x;
  for (; i < n; i += stride) dst[i] = f2bf(src[i]);
}

__global__ __launch_bounds__(256) void final_add_kernel(
    const float* __restrict__ h, const float* __restrict__ slots,
    float* __restrict__ out) {
  size_t i = (size_t)blockIdx.x * blockDim.x + threadIdx.x;
  size_t stride = (size_t)gridDim.x * blockDim.x;
  for (; i < (size_t)BS_ * H_; i += stride) {
    size_t t = i / H_;
    out[i] = h[i] + slots[i + t * H_] + slots[i + t * H_ + H_];
  }
}

// ---------------------------------------------------------------------------
extern "C" void kernel_launch(void* const* d_in, const int* in_sizes, int n_in,
                              void* d_out, int out_size, void* d_ws, size_t ws_size,
                              hipStream_t stream) {
  (void)in_sizes; (void)n_in; (void)out_size; (void)ws_size;
  const float* hidden = (const float*)d_in[0];
  const int*   posid  = (const int*)d_in[1];
  const float* wq     = (const float*)d_in[2];
  const float* wk     = (const float*)d_in[3];
  const float* wv     = (const float*)d_in[4];
  const float* wo     = (const float*)d_in[5];
  const float* gatew  = (const float*)d_in[6];
  const float* w1     = (const float*)d_in[7];
  const float* w3     = (const float*)d_in[8];
  const float* w2     = (const float*)d_in[9];
  const float* ln1    = (const float*)d_in[10];
  const float* ln2    = (const float*)d_in[11];

  float* hidden_out = (float*)d_out;                        // (B,S,H)
  float* logits_out = (float*)d_out + (size_t)BS_ * H_;     // (B,S,E)

  // workspace carve-up (256B aligned, deterministic)
  char* base = (char*)d_ws;
  size_t off = 0;
  auto alloc = [&](size_t bytes) -> void* {
    void* p = base + off;
    off = (off + bytes + 255) & ~(size_t)255;
    return p;
  };
  __bf16* xb   = (__bf16*)alloc((size_t)BS_ * H_ * 2);
  __bf16* wqb  = (__bf16*)alloc((size_t)H_ * NH_ * HD_ * 2);
  __bf16* wkb  = (__bf16*)alloc((size_t)H_ * NKV_ * HD_ * 2);
  __bf16* wvb  = (__bf16*)alloc((size_t)H_ * NKV_ * HD_ * 2);
  __bf16* wob  = (__bf16*)alloc((size_t)NH_ * HD_ * H_ * 2);
  __bf16* w1b  = (__bf16*)alloc((size_t)E_ * H_ * F_ * 2);
  __bf16* w3b  = (__bf16*)alloc((size_t)E_ * H_ * F_ * 2);
  __bf16* w2b  = (__bf16*)alloc((size_t)E_ * F_ * H_ * 2);
  float*  qf   = (float*)alloc((size_t)BS_ * NH_ * HD_ * 4);
  float*  kf   = (float*)alloc((size_t)BS_ * NKV_ * HD_ * 4);
  float*  vf   = (float*)alloc((size_t)BS_ * NKV_ * HD_ * 4);
  __bf16* qb   = (__bf16*)alloc((size_t)BS_ * NH_ * HD_ * 2);
  __bf16* kb   = (__bf16*)alloc((size_t)BS_ * NKV_ * HD_ * 2);
  __bf16* vb   = (__bf16*)alloc((size_t)BS_ * NKV_ * HD_ * 2);
  __bf16* ab   = (__bf16*)alloc((size_t)BS_ * H_ * 2);
  float*  hbuf = (float*)alloc((size_t)BS_ * H_ * 4);
  float*  yf   = (float*)alloc((size_t)BS_ * H_ * 4);
  __bf16* yb   = (__bf16*)alloc((size_t)BS_ * H_ * 2);
  float*  topw = (float*)alloc((size_t)BS_ * 2 * 4);
  int*    topi = (int*)alloc((size_t)BS_ * 2 * 4);
  int*    cntc = (int*)alloc(16 * 4);           // [0..7]=counts, [8..15]=cursor
  int*    seg  = (int*)alloc((E_ + 1) * 4);
  int*    rmap = (int*)alloc((size_t)GROWS_ * 4);
  float*  rscl = (float*)alloc((size_t)GROWS_ * 4);
  int*    gsrc = (int*)alloc((size_t)GROWS_ * 4);
  __bf16* ygb  = (__bf16*)alloc((size_t)GROWS_ * H_ * 2);
  float*  gact = (float*)alloc((size_t)GROWS_ * F_ * 4);
  float*  uact = (float*)alloc((size_t)GROWS_ * F_ * 4);
  __bf16* tb   = (__bf16*)alloc((size_t)GROWS_ * F_ * 2);
  float*  slot = (float*)alloc((size_t)GROWS_ * H_ * 4);

  // 0) weight casts fp32 -> bf16 (one pass; amortizes WMMA's 8x rate gain)
  cast_f32_bf16<<<2048, 256, 0, stream>>>(wq, wqb, (size_t)H_ * NH_ * HD_);
  cast_f32_bf16<<<2048, 256, 0, stream>>>(wk, wkb, (size_t)H_ * NKV_ * HD_);
  cast_f32_bf16<<<2048, 256, 0, stream>>>(wv, wvb, (size_t)H_ * NKV_ * HD_);
  cast_f32_bf16<<<2048, 256, 0, stream>>>(wo, wob, (size_t)NH_ * HD_ * H_);
  cast_f32_bf16<<<4096, 256, 0, stream>>>(w1, w1b, (size_t)E_ * H_ * F_);
  cast_f32_bf16<<<4096, 256, 0, stream>>>(w3, w3b, (size_t)E_ * H_ * F_);
  cast_f32_bf16<<<4096, 256, 0, stream>>>(w2, w2b, (size_t)E_ * F_ * H_);

  // 1) RMSNorm1
  rmsnorm_kernel<<<BS_, 256, 0, stream>>>(hidden, ln1, xb, nullptr);

  // 2) Q/K/V projections (bf16 WMMA)
  gemm_bf16_wmma<<<dim3(NH_ * HD_ / 128, BS_ / 128), 256, 0, stream>>>(
      xb, H_, wqb, NH_ * HD_, qf, NH_ * HD_, BS_, NH_ * HD_, H_,
      nullptr, 0, nullptr, nullptr, nullptr, 0);
  gemm_bf16_wmma<<<dim3(NKV_ * HD_ / 128, BS_ / 128), 256, 0, stream>>>(
      xb, H_, wkb, NKV_ * HD_, kf, NKV_ * HD_, BS_, NKV_ * HD_, H_,
      nullptr, 0, nullptr, nullptr, nullptr, 0);
  gemm_bf16_wmma<<<dim3(NKV_ * HD_ / 128, BS_ / 128), 256, 0, stream>>>(
      xb, H_, wvb, NKV_ * HD_, vf, NKV_ * HD_, BS_, NKV_ * HD_, H_,
      nullptr, 0, nullptr, nullptr, nullptr, 0);

  // 3) RoPE + cast to bf16
  rope_cast_kernel<<<BS_, 256, 0, stream>>>(qf, kf, vf, posid, qb, kb, vb);

  // 4) sliding-window flash attention (WMMA)
  attn_kernel<<<dim3(S_ / 64, NH_, B_), 128, 0, stream>>>(qb, kb, vb, ab);

  // 5) O projection + residual
  gemm_bf16_wmma<<<dim3(H_ / 128, BS_ / 128), 256, 0, stream>>>(
      ab, NH_ * HD_, wob, H_, hbuf, H_, BS_, H_, NH_ * HD_,
      nullptr, 0, hidden, nullptr, nullptr, 0);

  // 6) RMSNorm2 (f32 for router + bf16 for MoE GEMMs)
  rmsnorm_kernel<<<BS_, 256, 0, stream>>>(hbuf, ln2, yb, yf);

  // 7) router logits (second output) + top-2 routing
  router_kernel<<<BS_, 256, 0, stream>>>(yf, gatew, logits_out);
  zero_ints<<<1, 64, 0, stream>>>(cntc, 16);
  topk_kernel<<<(BS_ + 255) / 256, 256, 0, stream>>>(logits_out, topw, topi, cntc);
  scan_kernel<<<1, 32, 0, stream>>>(cntc, seg);
  scatter_kernel<<<(BS_ + 255) / 256, 256, 0, stream>>>(
      topi, topw, seg, cntc + 8, rmap, rscl, gsrc);
  gather_rows<<<GROWS_, 256, 0, stream>>>(yb, gsrc, ygb);

  // 8) per-expert gathered GEMMs (device-side segment bounds; blocks early-exit)
  for (int e = 0; e < E_; ++e) {
    gemm_bf16_wmma<<<dim3(F_ / 128, BS_ / 128), 256, 0, stream>>>(
        ygb, H_, w1b + (size_t)e * H_ * F_, F_, gact, F_, BS_, F_, H_,
        seg, e, nullptr, nullptr, nullptr, 0);
    gemm_bf16_wmma<<<dim3(F_ / 128, BS_ / 128), 256, 0, stream>>>(
        ygb, H_, w3b + (size_t)e * H_ * F_, F_, uact, F_, BS_, F_, H_,
        seg, e, nullptr, nullptr, nullptr, 0);
  }
  silu_mul_kernel<<<4096, 256, 0, stream>>>(gact, uact, tb, (size_t)GROWS_ * F_);
  for (int e = 0; e < E_; ++e) {
    gemm_bf16_wmma<<<dim3(H_ / 128, BS_ / 128), 256, 0, stream>>>(
        tb, F_, w2b + (size_t)e * F_ * H_, H_, slot, H_, BS_, H_, F_,
        seg, e, nullptr, rscl, rmap, 0);
  }

  // 9) residual + combine the two expert slots per token
  final_add_kernel<<<2048, 256, 0, stream>>>(hbuf, slot, hidden_out);
}
